// Modular_Encoder_10058813407475
// MI455X (gfx1250) — matrix operand order repaired and checked by reading, not compiled, not run on hardware
//
#include <hip/hip_runtime.h>
#include <hip/hip_bf16.h>

// ---------------------------------------------------------------------------
// MI455X (gfx1250) modular-encoder.
// ~146 GFLOP of GEMM; bf16 WMMA (v_wmma_f32_16x16x32_bf16, f32 accum) is the
// right operating point (fp32 WMMA is 16x16x4 = 8x lower FLOP/instr).
// All GEMM operands pre-converted to bf16 once (x: 16MB, W: 8.5MB), policy
// aggregation materialized by a bandwidth-bound pass (~11us at 23.3 TB/s),
// so the GEMM hot loop is: async global->LDS b128 copies (double-buffered,
// ASYNCcnt-pipelined) + ds_load_b128 fragment loads + WMMA. Workspace: 288MB.
// ---------------------------------------------------------------------------

typedef __attribute__((ext_vector_type(16))) __bf16 v16bf;
typedef __attribute__((ext_vector_type(8)))  __bf16 v8bf;
typedef __attribute__((ext_vector_type(4)))  __bf16 v4bf;
typedef __attribute__((ext_vector_type(8)))  float  v8f;
typedef __attribute__((ext_vector_type(4)))  float  v4f;
typedef __attribute__((__vector_size__(16))) int    vs4i;   // builtin's b128 unit

namespace {
constexpr int UNITS = 8;
constexpr int NB    = 16384;   // batch
constexpr int KD    = 512;     // IN == H == 512 (contraction dim everywhere)
constexpr int NCV   = 62;      // n_class
constexpr int LDK   = 40;      // LDS row stride in bf16 (80B, 16B aligned)
}

#define AS1 __attribute__((address_space(1)))
#define AS3 __attribute__((address_space(3)))

// 16B global(bf16) -> LDS copy. CDNA5 async-to-LDS (ASYNCcnt); synchronous
// fallback keeps the file compiling on non-gfx1250 passes.
__device__ inline void async_cp16(const __bf16* g, __bf16* l) {
#if defined(__gfx1250__) && __has_builtin(__builtin_amdgcn_global_load_async_to_lds_b128)
  __builtin_amdgcn_global_load_async_to_lds_b128(
      (AS1 vs4i*)(uintptr_t)g, (AS3 vs4i*)(uintptr_t)l, 0, 0);
#else
  *(v8bf*)l = *(const v8bf*)g;
#endif
}

template <int N> __device__ inline void wait_async() {
#if defined(__gfx1250__)
# if __has_builtin(__builtin_amdgcn_s_wait_asynccnt)
  __builtin_amdgcn_s_wait_asynccnt(N);
# else
  asm volatile("s_wait_asynccnt %0" :: "n"(N) : "memory");
# endif
#endif
}

// A-fragment: 16x32 bf16: lanes 0-15 -> M=lane, K 0..15;
// lanes 16-31 -> M=lane-16, K 16..31. Two ds_load_b128.
__device__ inline v16bf ldfragA(const __bf16* As, int m0, int lane) {
  const __bf16* p = As + (m0 + (lane & 15)) * LDK;
  const int kb = (lane >> 4) << 4;
  v8bf lo = *(const v8bf*)(p + kb);
  v8bf hi = *(const v8bf*)(p + 8 + kb);
  return __builtin_shufflevector(lo, hi, 0,1,2,3,4,5,6,7,8,9,10,11,12,13,14,15);
}

// B-fragment: 32x16 bf16 from N-major LDS [n][k]: lanes 0-15 -> N=lane, K 0..15;
// lanes 16-31 -> N=lane-16, K 16..31. Two ds_load_b128.
__device__ inline v16bf ldfragB(const __bf16* Bs, int n0, int lane) {
  const __bf16* p = Bs + (n0 + (lane & 15)) * LDK + ((lane >> 4) << 4);
  v8bf lo = *(const v8bf*)(p);
  v8bf hi = *(const v8bf*)(p + 8);
  return __builtin_shufflevector(lo, hi, 0,1,2,3,4,5,6,7,8,9,10,11,12,13,14,15);
}

// C = act( A @ Wt[u]^T + bias[u] ).  A: bf16 [NB][KD].  Wt: bf16 [UNITS][Nw][KD]
// (N-major, pre-transposed, zero-padded).  128 x BN tile per 8-wave block,
// double-buffered LDS, async-copy pipelined over the K loop.
template <int BN, bool RELU, bool OUT_BF16, bool NCLIP>
__global__ __launch_bounds__(256)
void gemm_bf(const __bf16* __restrict__ A, const __bf16* __restrict__ Wt,
             const float* __restrict__ bias, void* __restrict__ Out,
             int Nw, int Nreal) {
  constexpr int BM   = 128;
  constexpr int FN   = BN / 32;            // N fragments per wave (4 or 2)
  constexpr int BSEG = BN * 4;             // 16B segments per B tile (512/256)
  constexpr int CPT  = 2 + BSEG / 256;     // async copies per thread per stage

  __shared__ __bf16 As[2 * BM * LDK];
  __shared__ __bf16 Bs[2 * BN * LDK];

  const int tid   = threadIdx.x;
  const int lane  = tid & 31;
  const int wid   = tid >> 5;
  const int waveM = wid & 3;
  const int waveN = wid >> 2;
  const int u     = blockIdx.z;
  const int mb    = blockIdx.y * BM;
  const int nb    = blockIdx.x * BN;

  const __bf16* Au = A + (size_t)mb * KD;
  const __bf16* Wu = Wt + (size_t)u * Nw * KD + (size_t)nb * KD;

  v8f acc[2][FN] = {};

  auto stage = [&](int k0, int lb) {
    __bf16* Ad = &As[lb * BM * LDK];
    __bf16* Bd = &Bs[lb * BN * LDK];
    {  // A segments tid and tid+256 (row = s/4, 8-elem quarter = s%4)
      int s = tid, r = s >> 2, q = s & 3;
      async_cp16(Au + (size_t)r * KD + k0 + q * 8, Ad + r * LDK + q * 8);
      s = tid + 256; r = s >> 2; q = s & 3;
      async_cp16(Au + (size_t)r * KD + k0 + q * 8, Ad + r * LDK + q * 8);
    }
    {  // B segments
      int s = tid, n = s >> 2, q = s & 3;
      async_cp16(Wu + (size_t)n * KD + k0 + q * 8, Bd + n * LDK + q * 8);
      if (BSEG > 256) {
        s = tid + 256; n = s >> 2; q = s & 3;
        async_cp16(Wu + (size_t)n * KD + k0 + q * 8, Bd + n * LDK + q * 8);
      }
    }
  };

  constexpr int NSTEP = KD / 32;
  stage(0, 0);
  for (int i = 0; i < NSTEP; ++i) {
    const int lb = i & 1;
    if (i + 1 < NSTEP) {       // prefetch next tile, then wait for current
      stage((i + 1) * 32, lb ^ 1);
      wait_async<CPT>();
    } else {
      wait_async<0>();
    }
    __syncthreads();

    const __bf16* Ab = &As[lb * BM * LDK];
    const __bf16* Bb = &Bs[lb * BN * LDK];
    v16bf a0 = ldfragA(Ab, waveM * 32, lane);
    v16bf a1 = ldfragA(Ab, waveM * 32 + 16, lane);
    v16bf bf_[FN];
#pragma unroll
    for (int n = 0; n < FN; ++n) bf_[n] = ldfragB(Bb, waveN * (BN / 2) + n * 16, lane);
#pragma unroll
    for (int n = 0; n < FN; ++n) {
      acc[0][n] = __builtin_amdgcn_wmma_f32_16x16x32_bf16(false, a0, false, bf_[n],
                                                          (short)0, acc[0][n], false, false);
      acc[1][n] = __builtin_amdgcn_wmma_f32_16x16x32_bf16(false, a1, false, bf_[n],
                                                          (short)0, acc[1][n], false, false);
    }
    __syncthreads();  // all waves done reading lb before it is restaged
  }

  // ---- epilogue ----
  const int half = lane >> 4;
  const int cn   = lane & 15;
#pragma unroll
  for (int fm = 0; fm < 2; ++fm) {
#pragma unroll
    for (int fn = 0; fn < FN; ++fn) {
      const int gc = nb + waveN * (BN / 2) + fn * 16 + cn;
      float bv = 0.f;
      if (!NCLIP || gc < Nreal) bv = bias[u * Nreal + gc];
#pragma unroll
      for (int i = 0; i < 8; ++i) {
        const int gr = mb + waveM * 32 + fm * 16 + half * 8 + i;
        float val = acc[fm][fn][i] + bv;
        if (RELU) val = val > 0.f ? val : 0.f;
        if constexpr (OUT_BF16)
          ((__bf16*)Out)[(size_t)u * NB * KD + (size_t)gr * KD + gc] = (__bf16)val;
        else
          ((float*)Out)[(size_t)u * NB * 64 + (size_t)gr * 64 + gc] = val;
      }
    }
  }
}

// a[t] = where(row>0, (P h)[t]/row[t], (P h)[t]) for all 8 targets; bf16 in/out.
__global__ __launch_bounds__(256)
void aggregate_k(const __bf16* __restrict__ h, const float* __restrict__ policy,
                 __bf16* __restrict__ a) {
  const size_t vec = (size_t)blockIdx.x * blockDim.x + threadIdx.x;  // NB*KD/8
  const size_t off = vec * 8;
  float coef[UNITS][UNITS];
#pragma unroll
  for (int t = 0; t < UNITS; ++t) {
    float rs = 0.f;
#pragma unroll
    for (int f = 0; f < UNITS; ++f) { coef[t][f] = policy[t * UNITS + f]; rs += coef[t][f]; }
    const float s = rs > 0.f ? 1.f / rs : 1.f;
#pragma unroll
    for (int f = 0; f < UNITS; ++f) coef[t][f] *= s;
  }
  float src[UNITS][8];
#pragma unroll
  for (int f = 0; f < UNITS; ++f) {
    v8bf hv = *(const v8bf*)(h + (size_t)f * NB * KD + off);
#pragma unroll
    for (int i = 0; i < 8; ++i) src[f][i] = (float)hv[i];
  }
#pragma unroll
  for (int t = 0; t < UNITS; ++t) {
    v8bf o;
#pragma unroll
    for (int i = 0; i < 8; ++i) {
      float s = 0.f;
#pragma unroll
      for (int f = 0; f < UNITS; ++f) s += coef[t][f] * src[f][i];
      o[i] = (__bf16)s;
    }
    *(v8bf*)(a + (size_t)t * NB * KD + off) = o;
  }
}

// x f32 -> bf16, 4 elems/thread.
__global__ __launch_bounds__(256)
void cvt_x_k(const float* __restrict__ x, __bf16* __restrict__ xb) {
  const size_t i = ((size_t)blockIdx.x * blockDim.x + threadIdx.x) * 4;
  v4f v = *(const v4f*)(x + i);
  v4bf o;
  o[0] = (__bf16)v[0]; o[1] = (__bf16)v[1]; o[2] = (__bf16)v[2]; o[3] = (__bf16)v[3];
  *(v4bf*)(xb + i) = o;
}

// Wt[u][n][k] = (n < Nsrc) ? (bf16)W[u][k][n] : 0   (transpose + pad), 4 k/thread.
__global__ __launch_bounds__(256)
void cvt_w_k(const float* __restrict__ W, __bf16* __restrict__ Wt, int Nsrc, int Npad) {
  const int idx = blockIdx.x * blockDim.x + threadIdx.x;  // UNITS*Npad*KD/4
  const int kq = idx % (KD / 4);
  const int n  = (idx / (KD / 4)) % Npad;
  const int u  = idx / ((KD / 4) * Npad);
  v4bf o;
#pragma unroll
  for (int i = 0; i < 4; ++i) {
    float wv = 0.f;
    if (n < Nsrc) wv = W[((size_t)u * KD + kq * 4 + i) * Nsrc + n];
    o[i] = (__bf16)wv;
  }
  *(v4bf*)(Wt + ((size_t)u * Npad + n) * KD + kq * 4) = o;
}

// out = (out + h2[j]*pf[j]) * inv, j = 0..7 (exact reference recurrence).
__global__ __launch_bounds__(256)
void finalize_k(const float* __restrict__ h2, const float* __restrict__ pf,
                float* __restrict__ out) {
  const int idx = blockIdx.x * blockDim.x + threadIdx.x;
  if (idx >= NB * NCV) return;
  const int b = idx / NCV;
  const int c = idx - b * NCV;
  float p[UNITS];
  float d = 0.f;
#pragma unroll
  for (int j = 0; j < UNITS; ++j) { p[j] = pf[j]; d += p[j]; }
  const float inv = d > 0.f ? 1.f / d : 1.f;
  float o = 0.f;
#pragma unroll
  for (int j = 0; j < UNITS; ++j)
    o = (o + h2[(size_t)j * NB * 64 + (size_t)b * 64 + c] * p[j]) * inv;
  out[idx] = o;
}

extern "C" void kernel_launch(void* const* d_in, const int* in_sizes, int n_in,
                              void* d_out, int out_size, void* d_ws, size_t ws_size,
                              hipStream_t stream) {
  (void)in_sizes; (void)n_in; (void)out_size; (void)ws_size;
  const float* x  = (const float*)d_in[0];
  const float* p1 = (const float*)d_in[1];
  const float* p2 = (const float*)d_in[2];
  const float* pf = (const float*)d_in[3];
  const float* W0 = (const float*)d_in[4];
  const float* b0 = (const float*)d_in[5];
  const float* W1 = (const float*)d_in[6];
  const float* b1 = (const float*)d_in[7];
  const float* W2 = (const float*)d_in[8];
  const float* b2 = (const float*)d_in[9];

  // Workspace (288 MB):
  //   [0,4MB)        Wt0 bf16 [8][512][512]  (N-major)
  //   [4MB,8MB)      Wt1 bf16 [8][512][512]
  //   [8MB,8.5MB)    Wt2 bf16 [8][64][512]   (rows 62,63 zero)
  //   [16MB,32MB)    xb  bf16 [16384][512]
  //   [32MB,160MB)   slot0: h0 -> h1 -> h2(f32, 32MB)
  //   [160MB,288MB)  slot1: a1 -> a2
  char* ws = (char*)d_ws;
  __bf16* Wt0 = (__bf16*)ws;
  __bf16* Wt1 = Wt0 + (size_t)UNITS * KD * KD;
  __bf16* Wt2 = Wt1 + (size_t)UNITS * KD * KD;
  __bf16* xb  = (__bf16*)(ws + ((size_t)16 << 20));
  __bf16* s0  = (__bf16*)(ws + ((size_t)32 << 20));
  __bf16* s1  = (__bf16*)(ws + ((size_t)160 << 20));
  float*  h2  = (float*)s0;
  float*  out = (float*)d_out;

  dim3 blk(256);
  // prep: bf16 conversions (x) and weight transpose/convert/pad
  cvt_x_k<<<(NB * KD / 4) / 256, blk, 0, stream>>>(x, xb);
  cvt_w_k<<<(UNITS * 512 * KD / 4) / 256, blk, 0, stream>>>(W0, Wt0, 512, 512);
  cvt_w_k<<<(UNITS * 512 * KD / 4) / 256, blk, 0, stream>>>(W1, Wt1, 512, 512);
  cvt_w_k<<<(UNITS * 64 * KD / 4) / 256, blk, 0, stream>>>(W2, Wt2, NCV, 64);

  // L0: h0[u] = relu(xb @ Wt0[u]^T + b0[u])
  gemm_bf<128, true, true, false>
      <<<dim3(KD / 128, NB / 128, UNITS), blk, 0, stream>>>(xb, Wt0, b0, s0, 512, 512);
  // a1 = aggregate(policy1, h0)
  aggregate_k<<<(NB * KD / 8) / 256, blk, 0, stream>>>(s0, p1, s1);
  // L1: h1[u] = relu(a1[0] @ Wt1[u]^T + b1[u])   (h1 overwrites h0)
  gemm_bf<128, true, true, false>
      <<<dim3(KD / 128, NB / 128, UNITS), blk, 0, stream>>>(s1, Wt1, b1, s0, 512, 512);
  // a2 = aggregate(policy2, h1)
  aggregate_k<<<(NB * KD / 8) / 256, blk, 0, stream>>>(s0, p2, s1);
  // L2: h2[u] = a2[0] @ Wt2[u]^T + b2[u]   (N padded to 64; h2 overlays slot0)
  gemm_bf<64, false, false, true>
      <<<dim3(1, NB / 128, UNITS), blk, 0, stream>>>(s1, Wt2, b2, h2, 64, NCV);
  // final gated reduction over units
  finalize_k<<<(NB * NCV + 255) / 256, blk, 0, stream>>>(h2, pf, out);
}
